// FraudGNN_20710332301996
// MI455X (gfx1250) — compile-verified, matrix operand order checked by reading, not run on hardware
//
#include <hip/hip_runtime.h>
#include <hip/hip_bf16.h>

// ---------------------------------------------------------------------------
// 2-layer GCN forward for MI455X (gfx1250, wave32).
//   dinv = (deg+selfloop)^-1/2
//   xb   = bf16(X) row-major; w1p = W1 pre-swizzled into WMMA B-fragments
//   h1   = X @ W1 (WMMA bf16->f32); agg1 init fused: agg1 = h1*dinv^2
//   agg1 += scatter(norm * h1[src])        (f32 atomics, L2-resident)
//   h2   = relu(agg1+b1) @ W2              (wave-per-node, wave32 reduction)
//   agg2 = self + scatter(norm * h2) ; +b2 ; log_softmax
// ---------------------------------------------------------------------------

typedef __attribute__((ext_vector_type(16))) __bf16     v16bf;
typedef __attribute__((ext_vector_type(8)))  float      v8f;
typedef __attribute__((ext_vector_type(8)))  unsigned   v8u;

union FragAB { v8u u; v16bf b; };

__device__ __forceinline__ unsigned bf16_rne(float f) {
    unsigned u = __float_as_uint(f);
    u += 0x7FFFu + ((u >> 16) & 1u);          // round-to-nearest-even
    return u >> 16;
}
__device__ __forceinline__ unsigned pack2bf(float lo, float hi) {
    return bf16_rne(lo) | (bf16_rne(hi) << 16);
}

// ---------------- degree / dinv --------------------------------------------
__global__ void k_deg_init(float* dinv, int n) {
    int i = blockIdx.x * blockDim.x + threadIdx.x;
    if (i < n) dinv[i] = 1.0f;                 // self-loop
}
__global__ void k_deg_edges(const int* __restrict__ dst, float* dinv, int e) {
    int i = blockIdx.x * blockDim.x + threadIdx.x;
    if (i < e) atomicAdd(&dinv[dst[i]], 1.0f);
}
__global__ void k_deg_rsqrt(float* dinv, int n) {
    int i = blockIdx.x * blockDim.x + threadIdx.x;
    if (i < n) dinv[i] = rsqrtf(dinv[i]);
}

// ---------------- precision / layout prep ----------------------------------
// xb[i] = packed bf16 pair of (x[2i], x[2i+1]); row-major [N][64] dwords.
__global__ void k_pack_x(const float* __restrict__ x, unsigned* __restrict__ xb,
                         long long n_pairs) {
    long long i = (long long)blockIdx.x * blockDim.x + threadIdx.x;
    if (i < n_pairs) {
        float2 v = ((const float2*)x)[i];
        xb[i] = pack2bf(v.x, v.y);
    }
}
// w1p laid out exactly as the GEMM's B fragment: [ntile][kstep][lane][vgpr]
__global__ void k_pack_w1(const float* __restrict__ W1, unsigned* __restrict__ w1p) {
    int tid = blockIdx.x * blockDim.x + threadIdx.x;    // 8*4*32*8 = 8192
    if (tid >= 8192) return;
    int v    = tid & 7;
    int lane = (tid >> 3) & 31;
    int ks   = (tid >> 8) & 3;
    int nt   = tid >> 10;
    int k    = ks * 32 + ((v >> 2) << 4) + ((lane >> 4) << 3) + ((v & 3) << 1);
    int ncol = nt * 16 + (lane & 15);
    w1p[tid] = pack2bf(W1[(size_t)k * 128 + ncol], W1[(size_t)(k + 1) * 128 + ncol]);
}

// ---------------- layer-1 GEMM: h1 = X @ W1 via WMMA bf16 ------------------
// Block = 256 threads = 8 waves; block owns a 16-row M tile, wave w owns the
// 16-col N tile. All operands pre-packed -> inner loop is pure b128 + wmma.
// Epilogue stores h1 AND the self-loop-initialized agg1 = h1*dinv^2; bounds
// check hoisted to one block-level test so the common path is branch-free.
__global__ void k_gemm1_wmma(const unsigned* __restrict__ xb,   // [N][64]
                             const unsigned* __restrict__ w1p,  // [8][4][32][8]
                             const float* __restrict__ dinv,
                             float* __restrict__ h1,
                             float* __restrict__ agg1,
                             int n_nodes) {
    const int lane = threadIdx.x & 31;
    const int wid  = threadIdx.x >> 5;         // 0..7 : N tile
    const int m0   = blockIdx.x * 16;
    const int mr   = lane & 15;
    const int half = lane >> 4;

    int row = m0 + mr;
    if (row >= n_nodes) row = n_nodes - 1;
    const unsigned* xr = xb + (size_t)row * 64;     // dwords (bf16 pairs)
    const v8u* bfrag = (const v8u*)w1p;

    v8f acc = {};
#pragma unroll
    for (int ks = 0; ks < 4; ++ks) {
        FragAB a, b;
        // A: VGPRs 0..3 = K pairs ks*16 + half*4 + {0..3}; VGPRs 4..7 = +8
        uint4 alo = *(const uint4*)(xr + ks * 16 + half * 4);
        uint4 ahi = *(const uint4*)(xr + ks * 16 + 8 + half * 4);
        a.u = (v8u){alo.x, alo.y, alo.z, alo.w, ahi.x, ahi.y, ahi.z, ahi.w};
        b.u = bfrag[(wid * 4 + ks) * 32 + lane];     // 32B contiguous per lane
        acc = __builtin_amdgcn_wmma_f32_16x16x32_bf16(
                  false, a.b, false, b.b, (short)0, acc, false, false);
    }
    const int ncol  = wid * 16 + mr;
    const int mbase = m0 + (half << 3);              // C/D: M = v + 8*half
    if (m0 + 16 <= n_nodes) {                        // full tile: branch-free
        float*       hp = h1   + (size_t)mbase * 128 + ncol;
        float*       ap = agg1 + (size_t)mbase * 128 + ncol;
        const float4 d0 = *(const float4*)(dinv + mbase);
        const float4 d1 = *(const float4*)(dinv + mbase + 4);
        const float  dv[8] = {d0.x, d0.y, d0.z, d0.w, d1.x, d1.y, d1.z, d1.w};
#pragma unroll
        for (int v = 0; v < 8; ++v) {
            float val = acc[v];
            hp[(size_t)v * 128] = val;
            ap[(size_t)v * 128] = val * dv[v] * dv[v];
        }
    } else {                                         // tail tile
#pragma unroll
        for (int v = 0; v < 8; ++v) {
            const int m = mbase + v;
            if (m < n_nodes) {
                float val = acc[v];
                float di  = dinv[m];
                h1  [(size_t)m * 128 + ncol] = val;
                agg1[(size_t)m * 128 + ncol] = val * di * di;
            }
        }
    }
}

// ---------------- layer-1 edge scatter -------------------------------------
// one wave per edge; lane carries 4 channels (float4, coalesced 512B/wave).
// src/dst are wave-uniform: readfirstlane -> SALU norm + SGPR-base addressing.
__global__ void k_scatter1(const float* __restrict__ h1,
                           const int* __restrict__ src, const int* __restrict__ dst,
                           const float* __restrict__ dinv,
                           float* __restrict__ agg1, int n_edges) {
    int e    = (blockIdx.x * blockDim.x + threadIdx.x) >> 5;
    int lane = threadIdx.x & 31;
    if (e >= n_edges) return;
    int s = __builtin_amdgcn_readfirstlane(src[e]);
    int d = __builtin_amdgcn_readfirstlane(dst[e]);
    float norm = dinv[s] * dinv[d];
    float4 v = ((const float4*)(h1 + (size_t)s * 128))[lane];
    float* out = agg1 + (size_t)d * 128 + lane * 4;
    atomicAdd(out + 0, v.x * norm);
    atomicAdd(out + 1, v.y * norm);
    atomicAdd(out + 2, v.z * norm);
    atomicAdd(out + 3, v.w * norm);
}

// ---------------- layer 2: h2 = relu(agg1+b1) @ W2 (wave per node) ---------
__global__ void k_layer2(const float* __restrict__ agg1, const float* __restrict__ b1,
                         const float* __restrict__ W2, float* __restrict__ h2,
                         int n_nodes) {
    int node = (blockIdx.x * blockDim.x + threadIdx.x) >> 5;
    int lane = threadIdx.x & 31;
    if (node >= n_nodes) return;
    float4 v  = ((const float4*)(agg1 + (size_t)node * 128))[lane];
    float4 bb = ((const float4*)b1)[lane];
    float t0 = fmaxf(v.x + bb.x, 0.f), t1 = fmaxf(v.y + bb.y, 0.f);
    float t2 = fmaxf(v.z + bb.z, 0.f), t3 = fmaxf(v.w + bb.w, 0.f);
    int c = lane * 4;
    const float2* w2 = (const float2*)W2;              // [128] x float2
    float2 w0 = w2[c], w1 = w2[c + 1], w2v = w2[c + 2], w3 = w2[c + 3];
    float s0 = t0 * w0.x + t1 * w1.x + t2 * w2v.x + t3 * w3.x;
    float s1 = t0 * w0.y + t1 * w1.y + t2 * w2v.y + t3 * w3.y;
#pragma unroll
    for (int off = 16; off > 0; off >>= 1) {           // wave32 reduction
        s0 += __shfl_down(s0, off, 32);
        s1 += __shfl_down(s1, off, 32);
    }
    if (lane == 0) { h2[node * 2] = s0; h2[node * 2 + 1] = s1; }
}

// ---------------- layer-2 aggregation + log_softmax ------------------------
__global__ void k_self2(const float* __restrict__ h2, const float* __restrict__ dinv,
                        float* __restrict__ agg2, int n_nodes) {
    int i = blockIdx.x * blockDim.x + threadIdx.x;     // over 2N
    if (i < n_nodes * 2) {
        float di = dinv[i >> 1];
        agg2[i] = h2[i] * di * di;                     // full overwrite (init)
    }
}
__global__ void k_scatter2(const float* __restrict__ h2,
                           const int* __restrict__ src, const int* __restrict__ dst,
                           const float* __restrict__ dinv,
                           float* __restrict__ agg2, int n_edges) {
    int e = blockIdx.x * blockDim.x + threadIdx.x;
    if (e >= n_edges) return;
    int s = src[e], d = dst[e];
    float norm = dinv[s] * dinv[d];
    float2 hv = ((const float2*)h2)[s];
    atomicAdd(&agg2[d * 2 + 0], hv.x * norm);
    atomicAdd(&agg2[d * 2 + 1], hv.y * norm);
}
__global__ void k_logsoftmax(const float* __restrict__ agg2, const float* __restrict__ b2,
                             float* __restrict__ out, int n_nodes) {
    int n = blockIdx.x * blockDim.x + threadIdx.x;
    if (n >= n_nodes) return;
    float2 a = ((const float2*)agg2)[n];
    float z0 = a.x + b2[0];
    float z1 = a.y + b2[1];
    float m  = fmaxf(z0, z1);
    float l  = m + __logf(__expf(z0 - m) + __expf(z1 - m));
    ((float2*)out)[n] = make_float2(z0 - l, z1 - l);
}

// ---------------------------------------------------------------------------
extern "C" void kernel_launch(void* const* d_in, const int* in_sizes, int n_in,
                              void* d_out, int out_size, void* d_ws, size_t ws_size,
                              hipStream_t stream) {
    const float* x   = (const float*)d_in[0];
    const int*   ei  = (const int*)  d_in[1];
    const float* W1  = (const float*)d_in[2];
    const float* b1  = (const float*)d_in[3];
    const float* W2  = (const float*)d_in[4];
    const float* b2  = (const float*)d_in[5];
    float*       out = (float*)d_out;

    const int n_nodes = in_sizes[0] / 128;   // 100000
    const int n_edges = in_sizes[1] / 2;     // 1600000
    const int* src = ei;
    const int* dst = ei + n_edges;

    // workspace (f32/u32 elems):
    //   dinv[N] | xb[N*64] | w1p[8192] | h1[N*128] | agg1[N*128] | h2[2N]
    // agg2 aliases h1 (h1 dead once scatter1 completes).
    float*    ws   = (float*)d_ws;
    float*    dinv = ws;
    unsigned* xb   = (unsigned*)(dinv + n_nodes);
    unsigned* w1p  = xb + (size_t)n_nodes * 64;
    float*    h1   = (float*)(w1p + 8192);
    float*    agg1 = h1 + (size_t)n_nodes * 128;
    float*    h2   = agg1 + (size_t)n_nodes * 128;
    float*    agg2 = h1;

    const int B = 256;
    const long long n_pairs = (long long)n_nodes * 64;
    // degrees
    k_deg_init <<<(n_nodes + B - 1) / B, B, 0, stream>>>(dinv, n_nodes);
    k_deg_edges<<<(n_edges + B - 1) / B, B, 0, stream>>>(dst, dinv, n_edges);
    k_deg_rsqrt<<<(n_nodes + B - 1) / B, B, 0, stream>>>(dinv, n_nodes);
    // operand prep
    k_pack_x <<<(unsigned)((n_pairs + B - 1) / B), B, 0, stream>>>(x, xb, n_pairs);
    k_pack_w1<<<8192 / B, B, 0, stream>>>(W1, w1p);
    // layer 1 (GEMM epilogue writes h1 and self-loop-initialized agg1)
    k_gemm1_wmma<<<(n_nodes + 15) / 16, B, 0, stream>>>(xb, w1p, dinv, h1, agg1, n_nodes);
    k_scatter1<<<((size_t)n_edges * 32 + B - 1) / B, B, 0, stream>>>(h1, src, dst, dinv, agg1, n_edges);
    // layer 2
    k_layer2  <<<((size_t)n_nodes * 32 + B - 1) / B, B, 0, stream>>>(agg1, b1, W2, h2, n_nodes);
    k_self2   <<<(n_nodes * 2 + B - 1) / B, B, 0, stream>>>(h2, dinv, agg2, n_nodes);
    k_scatter2<<<(n_edges + B - 1) / B, B, 0, stream>>>(h2, src, dst, dinv, agg2, n_edges);
    k_logsoftmax<<<(n_nodes + B - 1) / B, B, 0, stream>>>(agg2, b2, out, n_nodes);
}
